// KNN_itc_34711925686950
// MI455X (gfx1250) — matrix operand order repaired and checked by reading, not compiled, not run on hardware
//
#include <hip/hip_runtime.h>

typedef __attribute__((ext_vector_type(16))) _Float16 v16h;
typedef __attribute__((ext_vector_type(8)))  _Float16 v8h;
typedef __attribute__((ext_vector_type(8)))  float    v8f;

#define B_IMG   128
#define P_SP    441
#define NROWS   (B_IMG * P_SP)   /* 56448 */
#define CDIM    64
#define NWAY    50
#define MSUP    2205
#define MPAD    2208             /* multiple of 16 */
#define TPW     2                /* 16-row tiles per wave */
#define WAVES   4                /* waves per block sharing the B panel */
#define CHUNK_TILES 6            /* N-tiles per LDS chunk */
#define CHUNK_COLS  (CHUNK_TILES * 16)   /* 96 columns, 12 KB */
#define NCHUNK  (MPAD / CHUNK_COLS)      /* 23 */
#define LPT     6                /* async b128 loads per thread per chunk: 12288B/(128*16B) */
#define NEG_BIG (-3.0e38f)

// ---------------- preprocessing: normalize query descriptors -> f16 ----------
__global__ void qnorm_kernel(const float* __restrict__ q, _Float16* __restrict__ qn) {
  int row = blockIdx.x * blockDim.x + threadIdx.x;
  if (row >= NROWS) return;
  int b = row / P_SP, p = row % P_SP;
  const float* base = q + (size_t)b * (CDIM * P_SP) + p;
  float ss = 0.f;
#pragma unroll 8
  for (int c = 0; c < CDIM; ++c) { float v = base[(size_t)c * P_SP]; ss += v * v; }
  float r = rsqrtf(ss);
  _Float16* o = qn + (size_t)row * CDIM;
#pragma unroll 8
  for (int c = 0; c < CDIM; ++c) o[c] = (_Float16)(base[(size_t)c * P_SP] * r);
}

// ---- normalize support columns, transpose to [NWAY][MPAD][CDIM] f16 ---------
__global__ void snorm_kernel(const float* __restrict__ S, _Float16* __restrict__ snt) {
  int idx = blockIdx.x * blockDim.x + threadIdx.x;
  if (idx >= NWAY * MPAD) return;
  int j = idx / MPAD, m = idx % MPAD;
  _Float16* o = snt + (size_t)idx * CDIM;
  if (m >= MSUP) {
#pragma unroll 8
    for (int c = 0; c < CDIM; ++c) o[c] = (_Float16)0.0f;
    return;
  }
  const float* base = S + (size_t)j * (CDIM * MSUP) + m;
  float ss = 0.f;
#pragma unroll 8
  for (int c = 0; c < CDIM; ++c) { float v = base[(size_t)c * MSUP]; ss += v * v; }
  float r = rsqrtf(ss);
#pragma unroll 8
  for (int c = 0; c < CDIM; ++c) o[c] = (_Float16)(base[(size_t)c * MSUP] * r);
}

__global__ void zero_out_kernel(float* __restrict__ out, int n) {
  int i = blockIdx.x * blockDim.x + threadIdx.x;
  if (i < n) out[i] = 0.f;
}

// branchless insert of x into sorted-descending triple (t0>=t1>=t2)
__device__ __forceinline__ void topk_insert(float& t0, float& t1, float& t2, float x) {
  float mn  = fminf(t0, x);
  t0        = fmaxf(t0, x);
  float mn1 = fminf(t1, mn);
  t1        = fmaxf(t1, mn);
  t2        = fmaxf(t2, mn1);
}

// merge two sorted-descending triples -> top-3 of union (in a)
__device__ __forceinline__ void topk_merge(float& a0, float& a1, float& a2,
                                           float b0, float b1, float b2) {
  float lo = fminf(a0, b0);
  float c0 = fmaxf(a0, b0);
  float h1 = fmaxf(a1, b1);
  float l1 = fminf(a1, b1);
  float c1 = fmaxf(lo, h1);
  float c2 = fmaxf(fminf(lo, h1), fmaxf(l1, fmaxf(a2, b2)));
  a0 = c0; a1 = c1; a2 = c2;
}

// issue async copy of one 12KB B chunk (linear) into LDS; 6 b128 per thread
__device__ __forceinline__ void issue_chunk(const _Float16* gsrc, const _Float16* ldst,
                                            int tid) {
  uint32_t lbase = (uint32_t)(uintptr_t)ldst;      // low 32 bits of flat-shared = LDS addr
  uint64_t gbase = (uint64_t)(uintptr_t)gsrc;
#pragma unroll
  for (int k = 0; k < LPT; ++k) {
    uint32_t loff = lbase + (uint32_t)((tid + k * 128) * 16);
    uint64_t ga   = gbase + (uint64_t)((tid + k * 128) * 16);
    asm volatile("global_load_async_to_lds_b128 %0, %1, off"
                 :: "v"(loff), "v"(ga) : "memory");
  }
}

// consume CHUNK_TILES 16-column tiles from LDS: WMMA + fused top-3 insert
template <bool MASK>
__device__ __forceinline__ void consume_chunk(const _Float16* buf, int col0,
                                              const v16h (&afrag)[TPW][2],
                                              int hv, int lq,
                                              float (&t3)[TPW][8][3]) {
#pragma unroll
  for (int u = 0; u < CHUNK_TILES; ++u) {
    // B fragment: lane -> column N = u*16+lq, K halves at hv*8 and 16+hv*8 (matches A)
    const _Float16* bcol = buf + (u * 16 + lq) * CDIM + hv * 8;
    v8h b0lo = *(const v8h*)(bcol);
    v8h b0hi = *(const v8h*)(bcol + 16);
    v8h b1lo = *(const v8h*)(bcol + 32);
    v8h b1hi = *(const v8h*)(bcol + 48);
    v16h bf0 = __builtin_shufflevector(b0lo, b0hi, 0,1,2,3,4,5,6,7,8,9,10,11,12,13,14,15);
    v16h bf1 = __builtin_shufflevector(b1lo, b1hi, 0,1,2,3,4,5,6,7,8,9,10,11,12,13,14,15);
    bool valid = true;
    if (MASK) valid = (col0 + u * 16 + lq) < MSUP;

    v8f acc[TPW];
#pragma unroll
    for (int t = 0; t < TPW; ++t) {           // independent chains interleaved
      v8f z = {};
      acc[t] = __builtin_amdgcn_wmma_f32_16x16x32_f16(false, afrag[t][0], false, bf0,
                                                      (short)0, z, false, false);
    }
#pragma unroll
    for (int t = 0; t < TPW; ++t)
      acc[t] = __builtin_amdgcn_wmma_f32_16x16x32_f16(false, afrag[t][1], false, bf1,
                                                      (short)0, acc[t], false, false);
#pragma unroll
    for (int t = 0; t < TPW; ++t)
#pragma unroll
      for (int i = 0; i < 8; ++i) {
        float x = valid ? acc[t][i] : NEG_BIG;
        topk_insert(t3[t][i][0], t3[t][i][1], t3[t][i][2], x);
      }
  }
}

// ------------------------- fused GEMM + top-3 kernel -------------------------
__global__ __launch_bounds__(WAVES * 32) void knn_itc_main(
    const _Float16* __restrict__ qn, const _Float16* __restrict__ snt,
    float* __restrict__ out) {
  __shared__ _Float16 sbuf[2][CHUNK_COLS * CDIM];   // 2 x 12KB double buffer

  const int tid  = threadIdx.x;
  const int wave = tid >> 5;
  const int lane = tid & 31;
  const int hv   = lane >> 4;
  const int lq   = lane & 15;
  const int tile0 = blockIdx.x * (WAVES * TPW) + wave * TPW;

  // A fragments (16x32 f16 ISA layout), both K halves, for TPW row-tiles
  v16h afrag[TPW][2];
#pragma unroll
  for (int t = 0; t < TPW; ++t) {
    const _Float16* qrow = qn + (size_t)((tile0 + t) * 16 + lq) * CDIM;
#pragma unroll
    for (int kf = 0; kf < 2; ++kf) {
      v8h lo = *(const v8h*)(qrow + kf * 32 + hv * 8);
      v8h hi = *(const v8h*)(qrow + kf * 32 + 16 + hv * 8);
      afrag[t][kf] = __builtin_shufflevector(lo, hi, 0,1,2,3,4,5,6,7,8,9,10,11,12,13,14,15);
    }
  }

  for (int j = 0; j < NWAY; ++j) {
    const _Float16* sj = snt + (size_t)j * MPAD * CDIM;

    float t3[TPW][8][3];
#pragma unroll
    for (int t = 0; t < TPW; ++t)
#pragma unroll
      for (int i = 0; i < 8; ++i) {
        t3[t][i][0] = NEG_BIG; t3[t][i][1] = NEG_BIG; t3[t][i][2] = NEG_BIG;
      }

    // software-pipelined, double-buffered async B staging through LDS
    issue_chunk(sj, &sbuf[0][0], tid);
    for (int c = 0; c < NCHUNK; ++c) {
      if (c + 1 < NCHUNK) {
        issue_chunk(sj + (size_t)(c + 1) * CHUNK_COLS * CDIM, &sbuf[(c + 1) & 1][0], tid);
        asm volatile("s_wait_asynccnt 6" ::: "memory");   // chunk c landed (in-order)
      } else {
        asm volatile("s_wait_asynccnt 0" ::: "memory");
      }
      __syncthreads();                                    // publish LDS to all waves
      if (c == NCHUNK - 1)
        consume_chunk<true>(&sbuf[c & 1][0], c * CHUNK_COLS, afrag, hv, lq, t3);
      else
        consume_chunk<false>(&sbuf[c & 1][0], c * CHUNK_COLS, afrag, hv, lq, t3);
      __syncthreads();                                    // seal buffer before re-fill
    }

    // merge per-lane top-3 across the 16 lanes of each half
#pragma unroll
    for (int d = 1; d <= 8; d <<= 1) {
#pragma unroll
      for (int t = 0; t < TPW; ++t)
#pragma unroll
        for (int i = 0; i < 8; ++i) {
          float b0 = __shfl_xor(t3[t][i][0], d, 32);
          float b1 = __shfl_xor(t3[t][i][1], d, 32);
          float b2 = __shfl_xor(t3[t][i][2], d, 32);
          topk_merge(t3[t][i][0], t3[t][i][1], t3[t][i][2], b0, b1, b2);
        }
    }

    if (lq == 0) {   // lanes 0/16 hold rows 8*hv .. 8*hv+7 of each tile
#pragma unroll
      for (int t = 0; t < TPW; ++t)
#pragma unroll
        for (int i = 0; i < 8; ++i) {
          int row = (tile0 + t) * 16 + hv * 8 + i;
          int b   = row / P_SP;
          float s = t3[t][i][0] + t3[t][i][1] + t3[t][i][2];
          atomicAdd(out + b * NWAY + j, s);
        }
    }
  }
}

extern "C" void kernel_launch(void* const* d_in, const int* in_sizes, int n_in,
                              void* d_out, int out_size, void* d_ws, size_t ws_size,
                              hipStream_t stream) {
  const float* q = (const float*)d_in[0];   // [128, 64, 21, 21]
  const float* S = (const float*)d_in[1];   // [50, 64, 2205]
  float* out = (float*)d_out;               // [128, 50]

  _Float16* qn  = (_Float16*)d_ws;                                     // 7.2 MB
  _Float16* snt = (_Float16*)((char*)d_ws + (size_t)NROWS * CDIM * 2); // 14.1 MB

  qnorm_kernel<<<(NROWS + 255) / 256, 256, 0, stream>>>(q, qn);
  snorm_kernel<<<(NWAY * MPAD + 255) / 256, 256, 0, stream>>>(S, snt);
  zero_out_kernel<<<(B_IMG * NWAY + 255) / 256, 256, 0, stream>>>(out, B_IMG * NWAY);
  knn_itc_main<<<NROWS / (16 * WAVES * TPW), WAVES * 32, 0, stream>>>(qn, snt, out);
}
